// LatePoolingCodeDecoder_23210003268000
// MI455X (gfx1250) — matrix operand order repaired and checked by reading, not compiled, not run on hardware
//
#include <hip/hip_runtime.h>
#include <cstddef>

typedef __bf16 bf16;
typedef __attribute__((ext_vector_type(16))) __bf16 v16bf;
typedef __attribute__((ext_vector_type(8)))  float  v8f;
typedef __attribute__((ext_vector_type(4)))  float  f32x4;

namespace {
constexpr int T_ = 256, B_ = 128, P_ = 5, L_ = 128, V_ = 64;
constexpr int H_ = 256, E_ = 256, IO_ = 512, CTX_ = 512;
constexpr int BP_ = B_ * P_;        // 640
constexpr int G_  = 4 * H_;         // 1024 gate columns
constexpr int K0_ = CTX_ + H_;      // 768  (A0 = [ctx | h0])
constexpr int K1_ = H_ + H_;        // 512  (A1 = [h0 | h1])
constexpr int KQ_ = H_;             // 256
constexpr int ROWS_PER_BLK = 2 * P_;        // 10 real rows (2 batch groups)
constexpr int NBLK = BP_ / ROWS_PER_BLK;    // 64 blocks
constexpr int NTHREADS = 512;               // 16 waves (wave32)

// ---- LDS layout (bytes) ----
constexpr int OFF_A0   = 0;                         // bf16 [16][768]
constexpr int OFF_A1   = OFF_A0 + 16 * K0_ * 2;     // bf16 [16][512]
constexpr int OFF_C0   = OFF_A1 + 16 * K1_ * 2;     // f32  [16][256]
constexpr int OFF_C1   = OFF_C0 + 16 * H_ * 4;      // f32  [16][256]
constexpr int OFF_QB   = OFF_C1 + 16 * H_ * 4;      // f32  [16][512]
constexpr int OFF_CTXF = OFF_QB + 16 * CTX_ * 4;    // f32  [16][512]
constexpr int OFF_AT   = OFF_CTXF + 16 * CTX_ * 4;  // f32  [16][128]
constexpr int OFF_PL   = OFF_AT + 16 * L_ * 4;      // f32  [2][768]
constexpr int SMEM_BYTES = OFF_PL + 2 * (CTX_ + H_) * 4;  // 153600
constexpr int ZERO_BYTES = OFF_QB;  // zero A0,A1,c0,c1 at start

// ---- workspace layout (bytes) ----
constexpr size_t WS_WP0   = 0;                                  // bf16 [1024][768]
constexpr size_t WS_WP1   = WS_WP0 + (size_t)G_ * K0_ * 2;      // bf16 [1024][512]
constexpr size_t WS_WPQ   = WS_WP1 + (size_t)G_ * K1_ * 2;      // bf16 [512][256]
constexpr size_t WS_BIAS1 = WS_WPQ + (size_t)CTX_ * KQ_ * 2;    // f32  [1024]
constexpr size_t WS_CPROJ = WS_BIAS1 + 4096;                    // f32  [64][1024]
constexpr size_t WS_BASE  = WS_CPROJ + (size_t)V_ * G_ * 4;     // f32  [640][1024]
}  // namespace

// ---------------- WMMA helpers (CDNA5 layouts, CDNA5 ISA) -------
__device__ __forceinline__ v8f wmma_bf16(v16bf a, v16bf b, v8f c) {
  return __builtin_amdgcn_wmma_f32_16x16x32_bf16(false, a, false, b, (short)0,
                                                 c, false, false);
}

// 16-bit A 16x32: lanes 0-15 -> row=lane, K {0..7,16..23}; lanes 16-31 -> K {8..15,24..31}
__device__ __forceinline__ v16bf load_a_frag(const bf16* base, int rowStride,
                                             int k0, int lane) {
  int m = lane & 15;
  int kb = (lane < 16) ? 0 : 8;
  const bf16* p = base + m * rowStride + k0 + kb;
  union { v16bf v; f32x4 q[2]; } u;
  u.q[0] = *reinterpret_cast<const f32x4*>(p);
  u.q[1] = *reinterpret_cast<const f32x4*>(p + 16);
  return u.v;
}

// 16-bit B 32x16: lanes 0-15 -> col=lane, K {0..7,16..23}; lanes 16-31 -> K {8..15,24..31}.
// wrow points at Wpack[n_global*stride + k0]; two 8-bf16 runs per lane.
__device__ __forceinline__ v16bf load_b_frag(const bf16* wrow, int lane) {
  const bf16* p = wrow + ((lane < 16) ? 0 : 8);
  union { v16bf v; f32x4 q[2]; } u;
  u.q[0] = *reinterpret_cast<const f32x4*>(p);
  u.q[1] = *reinterpret_cast<const f32x4*>(p + 16);
  return u.v;
}

__device__ __forceinline__ float sigf(float x) { return 1.0f / (1.0f + __expf(-x)); }
__device__ __forceinline__ float tanh_fast(float x) {
  return 1.0f - 2.0f / (__expf(2.0f * x) + 1.0f);
}

// ---------------- precompute kernels ----------------------------------------
__global__ void k_pack0(const float* __restrict__ Wih0,
                        const float* __restrict__ Whh0, bf16* __restrict__ Wp0) {
  int i = blockIdx.x * blockDim.x + threadIdx.x;
  if (i >= G_ * K0_) return;
  int n = i / K0_, k = i % K0_;
  float v = (k < CTX_) ? Wih0[(size_t)n * 1280 + (E_ + IO_) + k]
                       : Whh0[(size_t)n * H_ + (k - CTX_)];
  Wp0[i] = (bf16)v;
}

__global__ void k_pack1(const float* __restrict__ Wih1,
                        const float* __restrict__ Whh1, bf16* __restrict__ Wp1) {
  int i = blockIdx.x * blockDim.x + threadIdx.x;
  if (i >= G_ * K1_) return;
  int n = i / K1_, k = i % K1_;
  float v = (k < H_) ? Wih1[(size_t)n * H_ + k] : Whh1[(size_t)n * H_ + (k - H_)];
  Wp1[i] = (bf16)v;
}

__global__ void k_packq(const float* __restrict__ Wq, bf16* __restrict__ Wpq) {
  int i = blockIdx.x * blockDim.x + threadIdx.x;
  if (i >= CTX_ * KQ_) return;
  Wpq[i] = (bf16)Wq[i];
}

__global__ void k_bias1(const float* __restrict__ bih1,
                        const float* __restrict__ bhh1, float* __restrict__ b1) {
  int i = blockIdx.x * blockDim.x + threadIdx.x;
  if (i < G_) b1[i] = bih1[i] + bhh1[i];
}

__global__ void k_codeproj(const float* __restrict__ code_embed,
                           const float* __restrict__ Wih0,
                           float* __restrict__ cproj) {
  int i = blockIdx.x * blockDim.x + threadIdx.x;
  if (i >= V_ * G_) return;
  int v = i / G_, j = i % G_;
  const float* ce = code_embed + (size_t)v * E_;
  const float* w = Wih0 + (size_t)j * 1280;
  float s = 0.f;
  for (int e = 0; e < E_; ++e) s += ce[e] * w[e];
  cproj[i] = s;
}

__global__ void k_base(const float* __restrict__ io_embed,
                       const float* __restrict__ Wih0,
                       const float* __restrict__ bih0,
                       const float* __restrict__ bhh0, float* __restrict__ baseg) {
  int i = blockIdx.x * blockDim.x + threadIdx.x;
  if (i >= BP_ * G_) return;
  int n = i / G_, j = i % G_;
  const float* io = io_embed + (size_t)n * IO_;
  const float* w = Wih0 + (size_t)j * 1280 + E_;
  float s = bih0[j] + bhh0[j];
  for (int k = 0; k < IO_; ++k) s += io[k] * w[k];
  baseg[i] = s;
}

// ---------------- persistent decoder ----------------------------------------
__global__ __launch_bounds__(NTHREADS, 1) void decoder_persistent(
    const int* __restrict__ tokens, const int* __restrict__ lengths,
    const float* __restrict__ mem, const unsigned char* __restrict__ mask,
    const bf16* __restrict__ Wp0, const bf16* __restrict__ Wp1,
    const bf16* __restrict__ Wpq, const float* __restrict__ bias1,
    const float* __restrict__ cproj, const float* __restrict__ baseg,
    const float* __restrict__ bq, const float* __restrict__ Wout,
    const float* __restrict__ bout, float* __restrict__ out) {
  extern __shared__ char smem[];
  bf16* A0 = (bf16*)(smem + OFF_A0);      // [16][768] = [ctx | h0] bf16
  bf16* A1 = (bf16*)(smem + OFF_A1);      // [16][512] = [h0 | h1] bf16
  float* c0 = (float*)(smem + OFF_C0);    // [16][256]
  float* c1 = (float*)(smem + OFF_C1);    // [16][256]
  float* qb = (float*)(smem + OFF_QB);    // [16][512]
  float* ctxf = (float*)(smem + OFF_CTXF);// [16][512]
  float* at = (float*)(smem + OFF_AT);    // [16][128]
  float* pl = (float*)(smem + OFF_PL);    // [2][768]

  const int tid = threadIdx.x;
  const int lane = tid & 31;
  const int wv = tid >> 5;  // 16 waves
  const int blk = blockIdx.x;
  const int nbase = blk * ROWS_PER_BLK;  // global row base
  const int b0 = blk * 2, b1 = b0 + 1;
  const int lc = lane & 15;
  const int mhalf = (lane < 16) ? 0 : 8;

  // zero-init recurrent state (h=c=ctx=0)
  for (int i = tid; i < ZERO_BYTES / 16; i += NTHREADS)
    ((f32x4*)smem)[i] = f32x4{0.f, 0.f, 0.f, 0.f};
  __syncthreads();

  for (int t = 0; t < T_; ++t) {
    const int tok0 = tokens[t * B_ + b0];
    const int tok1 = tokens[t * B_ + b1];

    // ===== LSTM0 gates: [16x768]_bf16 @ Wp0^T -> [16x1024]_f32 =====
    {
      v8f acc[4] = {v8f{0,0,0,0,0,0,0,0}, v8f{0,0,0,0,0,0,0,0},
                    v8f{0,0,0,0,0,0,0,0}, v8f{0,0,0,0,0,0,0,0}};
      #pragma unroll 2
      for (int k0 = 0; k0 < K0_; k0 += 32) {
        v16bf a = load_a_frag(A0, K0_, k0, lane);
        #pragma unroll
        for (int u = 0; u < 4; ++u) {
          int n = (wv + 16 * u) * 16 + lc;  // i/f/g/o aligned quadruple
          v16bf b = load_b_frag(Wp0 + (size_t)n * K0_ + k0, lane);
          acc[u] = wmma_bf16(a, b, acc[u]);
        }
      }
      __syncthreads();
      const float* cp0r = cproj + (size_t)tok0 * G_;
      const float* cp1r = cproj + (size_t)tok1 * G_;
      #pragma unroll
      for (int r = 0; r < 8; ++r) {
        int M = r + mhalf;
        int hc = wv * 16 + lc;  // hidden column 0..255
        int nrow = nbase + ((M < ROWS_PER_BLK) ? M : ROWS_PER_BLK - 1);
        const float* cpr = (M >= P_ && M < ROWS_PER_BLK) ? cp1r : cp0r;
        const float* bs = baseg + (size_t)nrow * G_;
        float gi = acc[0][r] + bs[hc] + cpr[hc];
        float gf = acc[1][r] + bs[256 + hc] + cpr[256 + hc];
        float gg = acc[2][r] + bs[512 + hc] + cpr[512 + hc];
        float go = acc[3][r] + bs[768 + hc] + cpr[768 + hc];
        float cn = sigf(gf) * c0[M * H_ + hc] + sigf(gi) * tanh_fast(gg);
        c0[M * H_ + hc] = cn;
        bf16 hb = (bf16)(sigf(go) * tanh_fast(cn));
        A0[M * K0_ + CTX_ + hc] = hb;  // h0 for next-step LSTM0
        A1[M * K1_ + hc] = hb;         // h0 for LSTM1
      }
      __syncthreads();
    }

    // ===== LSTM1 gates: [16x512] @ Wp1^T -> [16x1024] =====
    {
      v8f acc[4] = {v8f{0,0,0,0,0,0,0,0}, v8f{0,0,0,0,0,0,0,0},
                    v8f{0,0,0,0,0,0,0,0}, v8f{0,0,0,0,0,0,0,0}};
      #pragma unroll 2
      for (int k0 = 0; k0 < K1_; k0 += 32) {
        v16bf a = load_a_frag(A1, K1_, k0, lane);
        #pragma unroll
        for (int u = 0; u < 4; ++u) {
          int n = (wv + 16 * u) * 16 + lc;
          v16bf b = load_b_frag(Wp1 + (size_t)n * K1_ + k0, lane);
          acc[u] = wmma_bf16(a, b, acc[u]);
        }
      }
      __syncthreads();
      #pragma unroll
      for (int r = 0; r < 8; ++r) {
        int M = r + mhalf;
        int hc = wv * 16 + lc;
        float gi = acc[0][r] + bias1[hc];
        float gf = acc[1][r] + bias1[256 + hc];
        float gg = acc[2][r] + bias1[512 + hc];
        float go = acc[3][r] + bias1[768 + hc];
        float cn = sigf(gf) * c1[M * H_ + hc] + sigf(gi) * tanh_fast(gg);
        c1[M * H_ + hc] = cn;
        A1[M * K1_ + H_ + hc] = (bf16)(sigf(go) * tanh_fast(cn));  // h1
      }
      __syncthreads();
    }

    // ===== q = h1 @ Wq^T + bq : [16x256] @ [256x512] =====
    {
      v8f acc[2] = {v8f{0,0,0,0,0,0,0,0}, v8f{0,0,0,0,0,0,0,0}};
      #pragma unroll 2
      for (int k0 = 0; k0 < KQ_; k0 += 32) {
        v16bf a = load_a_frag(A1 + H_, K1_, k0, lane);  // h1 slice of A1
        #pragma unroll
        for (int u = 0; u < 2; ++u) {
          int n = (2 * wv + u) * 16 + lc;
          v16bf b = load_b_frag(Wpq + (size_t)n * KQ_ + k0, lane);
          acc[u] = wmma_bf16(a, b, acc[u]);
        }
      }
      #pragma unroll
      for (int u = 0; u < 2; ++u)
        #pragma unroll
        for (int r = 0; r < 8; ++r) {
          int M = r + mhalf;
          int col = (2 * wv + u) * 16 + lc;
          qb[M * CTX_ + col] = acc[u][r] + bq[col];
        }
      __syncthreads();
    }

    // ===== attention: per real row (waves 0..9), mem L2-resident =====
    const int rr = wv;
    const size_t nrow = (size_t)(nbase + rr);
    const float* memrow = mem + nrow * (size_t)(L_ * CTX_);
    if (wv < ROWS_PER_BLK) {
      const unsigned char* mrow = mask + nrow * L_;
      const f32x4* qr4 = (const f32x4*)(qb + rr * CTX_);
      float sc[4];
      #pragma unroll
      for (int li = 0; li < 4; ++li) {
        int l = lane + 32 * li;
        const f32x4* ml = (const f32x4*)(memrow + (size_t)l * CTX_);
        float s = 0.f;
        #pragma unroll 4
        for (int d4 = 0; d4 < CTX_ / 4; ++d4) {
          f32x4 m4 = ml[d4], q4 = qr4[d4];
          s += m4.x * q4.x + m4.y * q4.y + m4.z * q4.z + m4.w * q4.w;
        }
        s *= 0.044194173824159216f;  // 1/sqrt(512)
        sc[li] = mrow[l] ? s : -1e9f;
      }
      float mx = fmaxf(fmaxf(sc[0], sc[1]), fmaxf(sc[2], sc[3]));
      #pragma unroll
      for (int off = 16; off > 0; off >>= 1) mx = fmaxf(mx, __shfl_xor(mx, off, 32));
      float se[4], ssum = 0.f;
      #pragma unroll
      for (int li = 0; li < 4; ++li) { se[li] = __expf(sc[li] - mx); ssum += se[li]; }
      #pragma unroll
      for (int off = 16; off > 0; off >>= 1) ssum += __shfl_xor(ssum, off, 32);
      float inv = 1.f / ssum;
      #pragma unroll
      for (int li = 0; li < 4; ++li) at[rr * L_ + lane + 32 * li] = se[li] * inv;
    }
    __syncthreads();
    if (wv < ROWS_PER_BLK) {
      float accd[16];
      #pragma unroll
      for (int j = 0; j < 16; ++j) accd[j] = 0.f;
      for (int l = 0; l < L_; ++l) {
        float a = at[rr * L_ + l];
        const float* ml = memrow + (size_t)l * CTX_;
        #pragma unroll
        for (int j = 0; j < 16; ++j) accd[j] += a * ml[lane + 32 * j];
      }
      #pragma unroll
      for (int j = 0; j < 16; ++j) {
        int d = lane + 32 * j;
        ctxf[rr * CTX_ + d] = accd[j];
        A0[rr * K0_ + d] = (bf16)accd[j];  // ctx for next-step LSTM0
      }
    }
    __syncthreads();

    // ===== max-pool over P and logits (tiny) =====
    for (int idx = tid; idx < 2 * (CTX_ + H_); idx += NTHREADS) {
      int g = idx / (CTX_ + H_), col = idx % (CTX_ + H_);
      float m = -3.0e38f;
      #pragma unroll
      for (int p = 0; p < P_; ++p) {
        int row = g * P_ + p;
        float v = (col < CTX_) ? ctxf[row * CTX_ + col]
                               : (float)A1[row * K1_ + H_ + (col - CTX_)];
        m = fmaxf(m, v);
      }
      pl[idx] = m;
    }
    __syncthreads();
    if (wv < 4) {
      int idx = wv * 32 + lane;  // 128 outputs = 2 groups x 64 vocab
      int g = idx >> 6, v = idx & 63;
      const f32x4* p4 = (const f32x4*)(pl + g * (CTX_ + H_));
      const f32x4* w4 = (const f32x4*)(Wout + (size_t)v * (CTX_ + H_));
      float s = bout[v];
      #pragma unroll 4
      for (int k = 0; k < (CTX_ + H_) / 4; ++k) {
        f32x4 a = p4[k], w = w4[k];
        s += a.x * w.x + a.y * w.y + a.z * w.z + a.w * w.w;
      }
      int b = b0 + g;
      s = (t < lengths[b]) ? s : 0.f;
      out[(size_t)t * (B_ * V_) + (size_t)b * V_ + v] = s;
    }
    __syncthreads();
  }
}

// ---------------- launch -----------------------------------------------------
extern "C" void kernel_launch(void* const* d_in, const int* in_sizes, int n_in,
                              void* d_out, int out_size, void* d_ws,
                              size_t ws_size, hipStream_t stream) {
  (void)in_sizes; (void)n_in; (void)out_size; (void)ws_size;
  const int* tokens = (const int*)d_in[0];
  const int* labels = (const int*)d_in[1];
  const int* lengths = (const int*)d_in[2];
  const float* io_embed = (const float*)d_in[3];
  const float* mem = (const float*)d_in[4];
  const unsigned char* mask = (const unsigned char*)d_in[5];
  const float* code_embed = (const float*)d_in[6];
  const float* Wih0 = (const float*)d_in[7];
  const float* Whh0 = (const float*)d_in[8];
  const float* bih0 = (const float*)d_in[9];
  const float* bhh0 = (const float*)d_in[10];
  const float* Wih1 = (const float*)d_in[11];
  const float* Whh1 = (const float*)d_in[12];
  const float* bih1 = (const float*)d_in[13];
  const float* bhh1 = (const float*)d_in[14];
  const float* Wq = (const float*)d_in[15];
  const float* bq = (const float*)d_in[16];
  const float* Wout = (const float*)d_in[17];
  const float* bout = (const float*)d_in[18];

  char* ws = (char*)d_ws;
  bf16* Wp0 = (bf16*)(ws + WS_WP0);
  bf16* Wp1 = (bf16*)(ws + WS_WP1);
  bf16* Wpq = (bf16*)(ws + WS_WPQ);
  float* bias1 = (float*)(ws + WS_BIAS1);
  float* cproj = (float*)(ws + WS_CPROJ);
  float* baseg = (float*)(ws + WS_BASE);
  float* out = (float*)d_out;

  k_pack0<<<(G_ * K0_ + 255) / 256, 256, 0, stream>>>(Wih0, Whh0, Wp0);
  k_pack1<<<(G_ * K1_ + 255) / 256, 256, 0, stream>>>(Wih1, Whh1, Wp1);
  k_packq<<<(CTX_ * KQ_ + 255) / 256, 256, 0, stream>>>(Wq, Wpq);
  k_bias1<<<(G_ + 255) / 256, 256, 0, stream>>>(bih1, bhh1, bias1);
  k_codeproj<<<(V_ * G_ + 255) / 256, 256, 0, stream>>>(code_embed, Wih0, cproj);
  k_base<<<(BP_ * G_ + 255) / 256, 256, 0, stream>>>(io_embed, Wih0, bih0, bhh0,
                                                     baseg);
  decoder_persistent<<<NBLK, NTHREADS, SMEM_BYTES, stream>>>(
      tokens, lengths, mem, mask, Wp0, Wp1, Wpq, bias1, cproj, baseg, bq, Wout,
      bout, out);
  // second tuple output: labels passthrough (raw int32 bits)
  hipMemcpyAsync((char*)d_out + (size_t)T_ * B_ * V_ * 4, labels,
                 (size_t)T_ * B_ * 4, hipMemcpyDeviceToDevice, stream);
}
